// VAE_5935644803114
// MI455X (gfx1250) — matrix operand order repaired and checked by reading, not compiled, not run on hardware
//
#include <hip/hip_runtime.h>
#include <hip/hip_bf16.h>
#include <stdint.h>

// ---------------- problem constants (match reference) ----------------
#define NODE   8
#define EMB    16
#define HID    128
#define KCODE  256
#define BATCH  4096
#define IMG    (3 * 96 * 96)      // 27648
#define EPS    1e-5f
#define BETA_C 0.25f

// ---------------- WMMA vector types ----------------
typedef __bf16          v16bf __attribute__((ext_vector_type(16)));
typedef float           v8f   __attribute__((ext_vector_type(8)));
typedef unsigned short  v16u  __attribute__((ext_vector_type(16)));
typedef unsigned short  u16x4 __attribute__((ext_vector_type(4)));

union ABCast { v16u u; v16bf b; };

__device__ inline unsigned short f2bf(float f) {
    union { float f; unsigned int u; } c; c.f = f;
    unsigned int u = c.u;
    u += 0x7FFFu + ((u >> 16) & 1u);      // round-to-nearest-even
    return (unsigned short)(u >> 16);
}

// =====================================================================
// Generic tiled GEMM:  C[M,N] = act(A[M,K] @ B[K,N] + bias[N])
// fp32 in / fp32 out, bf16 WMMA (v_wmma_f32_16x16x32_bf16) inner loop.
// Block = 256 threads = 8 waves in a 4(M) x 2(N) wave grid; each wave
// owns one 16-row A-fragment and TWO 16-col B-fragments/accumulators
// (2 WMMAs per K-step -> A-frag reuse x2).
// Tile: BM=64, BN=64, BK=32. fp32 -> bf16 conversion while staging.
// Full tiles take a branchless float4 load path; edge tiles fall back
// to guarded element loads. Next K-tile is prefetched at WGP scope
// under a uniform condition (no exec juggling in the hot loop).
// ACT: 0 = none, 1 = ELU, 2 = tanh, 3 = ReLU
// =====================================================================
template <int ACT>
__global__ __launch_bounds__(256)
void gemm_bf16_wmma(const float* __restrict__ A, const float* __restrict__ B,
                    const float* __restrict__ bias, float* __restrict__ C,
                    int M, int N, int K, int lda, int ldb, int ldc)
{
    // row stride 36 u16 = 72 B (multiple of 8 for b64 LDS stores)
    __shared__ unsigned short As [64][36];   // [m][k]
    __shared__ unsigned short BsT[64][36];   // [n][k] (transposed: K-pairs contiguous)

    const int tid  = threadIdx.x;
    const int lane = tid & 31;
    const int wid  = tid >> 5;
    const int wm   = wid >> 1;     // 0..3  (16-row slice)
    const int wn0  = wid & 1;      // 0..1  (32-col slice -> two 16-col tiles)
    const int hi   = lane >> 4;    // lane half (ISA 16-bit A/B layouts)
    const int l15  = lane & 15;

    const int m0 = blockIdx.y * 64;
    const int n0 = blockIdx.x * 64;

    const bool fullM = (m0 + 64 <= M);
    const bool fullN = (n0 + 64 <= N);

    v8f acc0 = {}, acc1 = {};

    for (int k0 = 0; k0 < K; k0 += 32) {
        const bool fullK = (k0 + 32 <= K);

        // ---- speculative prefetch of the NEXT K tile, uniform guard only;
        //      locality 3 -> WGP scope (pull into all cache levels)     ----
        if (fullM && fullN && (k0 + 64 <= K)) {
            {   // next B tile: 32 rows x 64 cols fp32 = 128 cachelines
                int r = tid >> 3, s = tid & 7;         // 32 x 8 threads
                __builtin_prefetch(&B[(long)(k0 + 32 + r) * ldb + (n0 + s * 8)], 0, 3);
            }
            {   // next A tile: 64 rows x 32 cols fp32 = 128 cachelines (2/row)
                int r = tid >> 2, s = tid & 3;         // 64 x 4 threads
                __builtin_prefetch(&A[(long)(m0 + r) * lda + (k0 + 32 + s * 8)], 0, 3);
            }
        }

        // ---- stage A tile 64x32 (fp32 -> bf16) ----
        if (fullM && fullK) {
            #pragma unroll
            for (int i = 0; i < 2; ++i) {
                int e = (tid + i * 256) * 4;       // 2048 elems / 256 thr / 4-wide
                int r = e >> 5, c = e & 31;
                float4 v = *reinterpret_cast<const float4*>(
                               &A[(long)(m0 + r) * lda + (k0 + c)]);
                u16x4 p = { f2bf(v.x), f2bf(v.y), f2bf(v.z), f2bf(v.w) };
                *reinterpret_cast<u16x4*>(&As[r][c]) = p;
            }
        } else {
            #pragma unroll
            for (int i = 0; i < 8; ++i) {
                int e = tid + i * 256;
                int r = e >> 5, c = e & 31;
                int gm = m0 + r, gk = k0 + c;
                float v = (gm < M && gk < K) ? A[(long)gm * lda + gk] : 0.0f;
                As[r][c] = f2bf(v);
            }
        }
        // ---- stage B tile 32x64 transposed ----
        if (fullN && fullK) {
            #pragma unroll
            for (int i = 0; i < 2; ++i) {
                int e = (tid + i * 256) * 4;
                int r = e >> 6, c = e & 63;        // r = k, c = n
                float4 v = *reinterpret_cast<const float4*>(
                               &B[(long)(k0 + r) * ldb + (n0 + c)]);
                BsT[c + 0][r] = f2bf(v.x);
                BsT[c + 1][r] = f2bf(v.y);
                BsT[c + 2][r] = f2bf(v.z);
                BsT[c + 3][r] = f2bf(v.w);
            }
        } else {
            #pragma unroll
            for (int i = 0; i < 8; ++i) {
                int e = tid + i * 256;
                int r = e >> 6, c = e & 63;
                int gk = k0 + r, gn = n0 + c;
                float v = (gk < K && gn < N) ? B[(long)gk * ldb + gn] : 0.0f;
                BsT[c][r] = f2bf(v);
            }
        }
        __syncthreads();

        // A fragment (ISA 7.12.2, 16-bit A 16x32):
        //  lanes 0-15: V0-3 = K0..7,  V4-7 = K16..23
        //  lanes16-31: V0-3 = K8..15, V4-7 = K24..31
        ABCast a, b0, b1;
        {
            const int mr = wm * 16 + l15;
            #pragma unroll
            for (int i = 0; i < 4; ++i) {
                int k = hi * 8 + 2 * i;
                a.u[2 * i]     = As[mr][k];
                a.u[2 * i + 1] = As[mr][k + 1];
                int k2 = 16 + k;
                a.u[8 + 2 * i]     = As[mr][k2];
                a.u[8 + 2 * i + 1] = As[mr][k2 + 1];
            }
            // B fragment (16-bit B 32x16): same K order as A (lanes 0-15 K0..7,K16..23; lanes 16-31 K8..15,K24..31)
            const int nc = wn0 * 32 + l15;
            #pragma unroll
            for (int v = 0; v < 8; ++v) {
                int k = (v < 4) ? (hi * 8 + 2 * v) : (16 + hi * 8 + 2 * (v - 4));
                b0.u[2 * v]     = BsT[nc][k];
                b0.u[2 * v + 1] = BsT[nc][k + 1];
                b1.u[2 * v]     = BsT[nc + 16][k];
                b1.u[2 * v + 1] = BsT[nc + 16][k + 1];
            }
        }
        acc0 = __builtin_amdgcn_wmma_f32_16x16x32_bf16(
                   false, a.b, false, b0.b, (short)0, acc0, false, false);
        acc1 = __builtin_amdgcn_wmma_f32_16x16x32_bf16(
                   false, a.b, false, b1.b, (short)0, acc1, false, false);
        __syncthreads();
    }

    // epilogue: C/D layout — VGPR r: lanes 0-15 -> M=r, lanes 16-31 -> M=8+r
    const int nA = n0 + wn0 * 32 + l15;
    const int nB = nA + 16;
    #pragma unroll
    for (int r = 0; r < 8; ++r) {
        int m = m0 + wm * 16 + hi * 8 + r;
        if (m < M) {
            if (nA < N) {
                float v = acc0[r] + bias[nA];
                if      (ACT == 1) v = v > 0.0f ? v : (__expf(v) - 1.0f);
                else if (ACT == 2) v = tanhf(v);
                else if (ACT == 3) v = v > 0.0f ? v : 0.0f;
                C[(long)m * ldc + nA] = v;
            }
            if (nB < N) {
                float v = acc1[r] + bias[nB];
                if      (ACT == 1) v = v > 0.0f ? v : (__expf(v) - 1.0f);
                else if (ACT == 2) v = tanhf(v);
                else if (ACT == 3) v = v > 0.0f ? v : 0.0f;
                C[(long)m * ldc + nB] = v;
            }
        }
    }
}

// =====================================================================
// BatchNorm (train mode): per-column mean / biased var over the batch.
// One block per column; fixed-order tree reduction -> deterministic.
// =====================================================================
__global__ __launch_bounds__(256)
void bn_stats(const float* __restrict__ h, float* __restrict__ mu,
              float* __restrict__ var)
{
    __shared__ float s1[256], s2[256];
    const int c = blockIdx.x;                 // 0..127
    float a = 0.0f, b = 0.0f;
    for (int r = threadIdx.x; r < BATCH; r += 256) {
        float v = h[(long)r * (NODE * EMB) + c];
        a += v; b += v * v;
    }
    s1[threadIdx.x] = a; s2[threadIdx.x] = b;
    __syncthreads();
    for (int s = 128; s > 0; s >>= 1) {
        if (threadIdx.x < s) {
            s1[threadIdx.x] += s1[threadIdx.x + s];
            s2[threadIdx.x] += s2[threadIdx.x + s];
        }
        __syncthreads();
    }
    if (threadIdx.x == 0) {
        float m = s1[0] / (float)BATCH;
        mu[c]  = m;
        var[c] = s2[0] / (float)BATCH - m * m;
    }
}

__global__ void bn_apply(const float* __restrict__ h, const float* __restrict__ mu,
                         const float* __restrict__ var, const float* __restrict__ gamma,
                         const float* __restrict__ beta, float* __restrict__ latent)
{
    int i = blockIdx.x * blockDim.x + threadIdx.x;
    if (i >= BATCH * NODE * EMB) return;
    int c = i & (NODE * EMB - 1);
    latent[i] = (h[i] - mu[c]) * rsqrtf(var[c] + EPS) * gamma[c] + beta[c];
}

// =====================================================================
// Per-node VQ. Codebook (16 KB raw fp32) + z vectors are staged into LDS
// with CDNA5 async global->LDS copies (ASYNCcnt), then one wave per
// sample strides over the 256 codes; butterfly argmin with smaller-index
// tie-break (matches jnp.argmin). Deterministic partial-SSE reduction.
// =====================================================================
__global__ __launch_bounds__(256)
void vq_kernel(const float* __restrict__ latent, const float* __restrict__ codebooks,
               float* __restrict__ exo, float* __restrict__ partials)
{
    __shared__ __align__(16) float cb[KCODE * EMB];   // 16 KB
    __shared__ __align__(16) float zs[64 * EMB];      // 4 KB
    __shared__ float ssum[8];
    const int j   = blockIdx.y;         // node
    const int s0  = blockIdx.x * 64;    // first sample of this block
    const int tid = threadIdx.x, lane = tid & 31, wid = tid >> 5;

    const float* cbsrc = codebooks + (long)j * KCODE * EMB;
#if defined(__gfx1250__)
    // async 16-byte copies: 1024 chunks, 4 per thread
    #pragma unroll
    for (int i = 0; i < 4; ++i) {
        int c4 = tid + i * 256;
        unsigned           ldsa = (unsigned)(uintptr_t)(&cb[c4 * 4]);
        unsigned long long ga   = (unsigned long long)(uintptr_t)(cbsrc + c4 * 4);
        asm volatile("global_load_async_to_lds_b128 %0, %1, off"
                     :: "v"(ldsa), "v"(ga) : "memory");
    }
    // async per-lane 4-byte gathers of the z block (64 samples x 16 dims)
    #pragma unroll
    for (int i = 0; i < 4; ++i) {
        int e = tid + i * 256;
        int s = e >> 4, d = e & 15;
        unsigned           ldsa = (unsigned)(uintptr_t)(&zs[e]);
        unsigned long long ga   = (unsigned long long)(uintptr_t)(
            latent + (long)(s0 + s) * (NODE * EMB) + j * EMB + d);
        asm volatile("global_load_async_to_lds_b32 %0, %1, off"
                     :: "v"(ldsa), "v"(ga) : "memory");
    }
    asm volatile("s_wait_asynccnt 0x0" ::: "memory");
#else
    for (int i = tid; i < KCODE * EMB; i += 256) cb[i] = cbsrc[i];
    for (int i = tid; i < 64 * EMB; i += 256) {
        int s = i >> 4, d = i & 15;
        zs[i] = latent[(long)(s0 + s) * (NODE * EMB) + j * EMB + d];
    }
#endif
    __syncthreads();

    float wsse = 0.0f;
    for (int t = 0; t < 8; ++t) {
        const int s = wid * 8 + t;
        const float* z = &zs[s * EMB];
        float best = 3.4e38f; int bidx = 0x7fffffff;
        for (int c = lane; c < KCODE; c += 32) {
            const float* cv = &cb[c * EMB];
            float d = 0.0f;
            #pragma unroll
            for (int e = 0; e < EMB; ++e) { float df = z[e] - cv[e]; d += df * df; }
            if (d < best) { best = d; bidx = c; }    // first-min within lane
        }
        #pragma unroll
        for (int o = 16; o > 0; o >>= 1) {
            float ob = __shfl_xor(best, o, 32);
            int   oi = __shfl_xor(bidx, o, 32);
            if (ob < best || (ob == best && oi < bidx)) { best = ob; bidx = oi; }
        }
        float df2 = 0.0f;
        if (lane < EMB) {
            float q  = cb[bidx * EMB + lane];
            exo[(long)(s0 + s) * (NODE * EMB) + j * EMB + lane] = q;
            float zz = z[lane];
            df2 = (q - zz) * (q - zz);
        }
        #pragma unroll
        for (int o = 16; o > 0; o >>= 1) df2 += __shfl_xor(df2, o, 32);
        if (lane == 0) wsse += df2;
    }
    if (lane == 0) ssum[wid] = wsse;
    __syncthreads();
    if (tid == 0) {
        float tsum = 0.0f;
        for (int w = 0; w < 8; ++w) tsum += ssum[w];
        partials[blockIdx.y * gridDim.x + blockIdx.x] = tsum;
    }
}

__global__ __launch_bounds__(256)
void vq_reduce(const float* __restrict__ partials, int n, float* __restrict__ out)
{
    __shared__ float s[256];
    float a = 0.0f;
    for (int i = threadIdx.x; i < n; i += 256) a += partials[i];
    s[threadIdx.x] = a;
    __syncthreads();
    for (int st = 128; st > 0; st >>= 1) {
        if (threadIdx.x < st) s[threadIdx.x] += s[threadIdx.x + st];
        __syncthreads();
    }
    if (threadIdx.x == 0)
        out[0] = s[0] * ((1.0f + BETA_C) / (float)(BATCH * EMB));
}

// label_hat[i, j] = hj[i] . align_w[j] + align_b[j]   (N=1 -> plain dots)
__global__ void labels_kernel(const float* __restrict__ hjbuf,
                              const float* __restrict__ align_w,
                              const float* __restrict__ align_b,
                              float* __restrict__ out)
{
    int i = blockIdx.x * blockDim.x + threadIdx.x;   // i = s*NODE + j
    if (i >= BATCH * NODE) return;
    int s = i >> 3, j = i & 7;
    const float* h = hjbuf + (long)j * BATCH * HID + (long)s * HID;
    const float* w = align_w + j * HID;
    float a = 0.0f;
    for (int k = 0; k < HID; ++k) a += h[k] * w[k];
    out[i] = a + align_b[j];
}

// =====================================================================
extern "C" void kernel_launch(void* const* d_in, const int* in_sizes, int n_in,
                              void* d_out, int out_size, void* d_ws, size_t ws_size,
                              hipStream_t stream)
{
    (void)in_sizes; (void)n_in; (void)out_size; (void)ws_size;

    const float* x         = (const float*)d_in[0];
    // d_in[1] = B_adj. tril(ones,-1)[:,j] selects nodes i>j, which are still
    // zero when node j is produced -> parents block is exactly 0 in the
    // reference; only shared_w rows [128:144) (the exo slice) matter.
    const float* enc_w1    = (const float*)d_in[2];
    const float* enc_b1    = (const float*)d_in[3];
    const float* enc_w2    = (const float*)d_in[4];
    const float* enc_b2    = (const float*)d_in[5];
    const float* enc_w3    = (const float*)d_in[6];
    const float* enc_b3    = (const float*)d_in[7];
    const float* bn_gamma  = (const float*)d_in[8];
    const float* bn_beta   = (const float*)d_in[9];
    const float* codebooks = (const float*)d_in[10];
    const float* shared_w  = (const float*)d_in[11];
    const float* shared_b  = (const float*)d_in[12];
    const float* npsem_w   = (const float*)d_in[13];
    const float* npsem_b   = (const float*)d_in[14];
    const float* align_w   = (const float*)d_in[15];
    const float* align_b   = (const float*)d_in[16];
    const float* dec_w1    = (const float*)d_in[17];
    const float* dec_b1    = (const float*)d_in[18];
    const float* dec_w2    = (const float*)d_in[19];
    const float* dec_b2    = (const float*)d_in[20];
    const float* dec_w3    = (const float*)d_in[21];
    const float* dec_b3    = (const float*)d_in[22];

    // d_out: causal [4096,8,16] | xhat [4096,96,96,3] | vq_loss | label_hat [4096,8]
    float* causal_out = (float*)d_out;
    float* xhat_out   = causal_out + (long)BATCH * NODE * EMB;
    float* vq_out     = xhat_out + (long)BATCH * IMG;
    float* label_out  = vq_out + 1;

    // workspace carve-up (~43 MB)
    float* w = (float*)d_ws;
    float* h1       = w;  w += (long)BATCH * 300;
    float* h2       = w;  w += (long)BATCH * 300;
    float* h3       = w;  w += (long)BATCH * NODE * EMB;
    float* latent   = w;  w += (long)BATCH * NODE * EMB;
    float* mu       = w;  w += NODE * EMB;
    float* var      = w;  w += NODE * EMB;
    float* exo      = w;  w += (long)BATCH * NODE * EMB;
    float* hjbuf    = w;  w += (long)NODE * BATCH * HID;
    float* d1       = w;  w += (long)BATCH * 300;
    float* d2       = w;  w += (long)BATCH * 300;
    float* partials = w;  w += (BATCH / 64) * NODE;

    const dim3 blk(256);
    auto gg = [](int M, int N) { return dim3((N + 63) / 64, (M + 63) / 64); };

    // ---- encoder ----
    gemm_bf16_wmma<1><<<gg(BATCH, 300), blk, 0, stream>>>(
        x,  enc_w1, enc_b1, h1, BATCH, 300, IMG, IMG, 300, 300);
    gemm_bf16_wmma<1><<<gg(BATCH, 300), blk, 0, stream>>>(
        h1, enc_w2, enc_b2, h2, BATCH, 300, 300, 300, 300, 300);
    gemm_bf16_wmma<0><<<gg(BATCH, NODE * EMB), blk, 0, stream>>>(
        h2, enc_w3, enc_b3, h3, BATCH, NODE * EMB, 300, 300, NODE * EMB, NODE * EMB);

    // ---- batchnorm (train mode) ----
    bn_stats<<<dim3(NODE * EMB), blk, 0, stream>>>(h3, mu, var);
    bn_apply<<<dim3((BATCH * NODE * EMB + 255) / 256), blk, 0, stream>>>(
        h3, mu, var, bn_gamma, bn_beta, latent);

    // ---- per-node vector quantization ----
    vq_kernel<<<dim3(BATCH / 64, NODE), blk, 0, stream>>>(latent, codebooks, exo, partials);
    vq_reduce<<<dim3(1), blk, 0, stream>>>(partials, (BATCH / 64) * NODE, vq_out);

    // ---- causal latents (parents block == 0 -> parallel over nodes) ----
    for (int j = 0; j < NODE; ++j) {
        // hj = relu(exo_j @ shared_w[j][128:144,:] + shared_b[j])   [4096,128], K=16
        gemm_bf16_wmma<3><<<gg(BATCH, HID), blk, 0, stream>>>(
            exo + j * EMB,
            shared_w + (long)j * (NODE * EMB + EMB) * HID + (long)(NODE * EMB) * HID,
            shared_b + j * HID,
            hjbuf + (long)j * BATCH * HID,
            BATCH, HID, EMB, NODE * EMB, HID, HID);
        // causal[:, j] = tanh(hj @ npsem_w[j] + npsem_b[j])  -> d_out column block
        gemm_bf16_wmma<2><<<gg(BATCH, EMB), blk, 0, stream>>>(
            hjbuf + (long)j * BATCH * HID,
            npsem_w + (long)j * HID * EMB,
            npsem_b + j * EMB,
            causal_out + j * EMB,
            BATCH, EMB, HID, HID, EMB, NODE * EMB);
    }
    labels_kernel<<<dim3((BATCH * NODE + 255) / 256), blk, 0, stream>>>(
        hjbuf, align_w, align_b, label_out);

    // ---- decoder ----
    gemm_bf16_wmma<1><<<gg(BATCH, 300), blk, 0, stream>>>(
        causal_out, dec_w1, dec_b1, d1, BATCH, 300, NODE * EMB, NODE * EMB, 300, 300);
    gemm_bf16_wmma<1><<<gg(BATCH, 300), blk, 0, stream>>>(
        d1, dec_w2, dec_b2, d2, BATCH, 300, 300, 300, 300, 300);
    gemm_bf16_wmma<2><<<gg(BATCH, IMG), blk, 0, stream>>>(
        d2, dec_w3, dec_b3, xhat_out, BATCH, IMG, 300, 300, IMG, IMG);
}
